// Block_23476291240450
// MI455X (gfx1250) — matrix operand order repaired and checked by reading, not compiled, not run on hardware
//
#include <hip/hip_runtime.h>
#include <hip/hip_bf16.h>
#include <cstdint>
#include <cstddef>

// Problem constants (from reference)
#define B_    32
#define S_    64
#define D_    256
#define H_    8
#define HD_   32
#define E_    8
#define F_    1024
#define LAT_  64
#define NTOK  2048           // B*S == B*LATENT
#define KBIG  16384          // S*D == LATENT*D

typedef __attribute__((ext_vector_type(16))) __bf16 v16bf;
typedef __attribute__((ext_vector_type(8)))  __bf16 v8bf;
typedef __attribute__((ext_vector_type(8)))  float  v8f;
typedef __attribute__((ext_vector_type(4)))  float  v4f;
typedef __attribute__((ext_vector_type(2)))  float  v2f;

// ---------------------------------------------------------------------------
// WMMA fragment loaders (wave32, 16x16x32 bf16).
// A (16xK, 16-bit): lanes 0-15 row m, K chunks [kb..kb+7] and [kb+16..kb+23],
//   kb = k0 + 8*(lane>>4).  B pre-transposed to [N,K]: lane's row n, same
//   K chunks [kb..kb+7] and [kb+16..kb+23].
// ---------------------------------------------------------------------------
__device__ __forceinline__ v16bf load_a_frag_bf16(const __bf16* A, int Kdim, int m, int kb) {
    const v8bf* p0 = (const v8bf*)(A + (size_t)m * Kdim + kb);
    const v8bf* p1 = (const v8bf*)(A + (size_t)m * Kdim + kb + 16);
    v8bf c0 = *p0, c1 = *p1;
    v16bf a;
#pragma unroll
    for (int i = 0; i < 8; ++i) { a[i] = c0[i]; a[i + 8] = c1[i]; }
    return a;
}

__device__ __forceinline__ v16bf load_a_frag_f32(const float* A, int Kdim, int m, int kb) {
    const v4f* p = (const v4f*)(A + (size_t)m * Kdim + kb);
    v4f f0 = p[0], f1 = p[1];
    const v4f* q = (const v4f*)(A + (size_t)m * Kdim + kb + 16);
    v4f f2 = q[0], f3 = q[1];
    v16bf a;
#pragma unroll
    for (int i = 0; i < 4; ++i) {
        a[i]      = (__bf16)f0[i];
        a[i + 4]  = (__bf16)f1[i];
        a[i + 8]  = (__bf16)f2[i];
        a[i + 12] = (__bf16)f3[i];
    }
    return a;
}

__device__ __forceinline__ v16bf load_b_frag(const __bf16* Bt, int Kdim, int n, int kb) {
    const v8bf* p = (const v8bf*)(Bt + (size_t)n * Kdim + kb);
    v8bf c0 = p[0], c1 = p[2];
    v16bf b;
#pragma unroll
    for (int i = 0; i < 8; ++i) { b[i] = c0[i]; b[i + 8] = c1[i]; }
    return b;
}

// ---------------------------------------------------------------------------
// RMSNorm over D=256: one block per row, 256 threads.
// ---------------------------------------------------------------------------
__global__ void rmsnorm_kernel(const float* __restrict__ x, const float* __restrict__ w,
                               float* __restrict__ y) {
    int row = blockIdx.x;
    int t   = threadIdx.x;
    float v = x[(size_t)row * D_ + t];
    __shared__ float red[256];
    red[t] = v * v;
    __syncthreads();
#pragma unroll
    for (int s = 128; s > 0; s >>= 1) {
        if (t < s) red[t] += red[t + s];
        __syncthreads();
    }
    float ms = red[0] * (1.0f / D_);
    y[(size_t)row * D_ + t] = v * rsqrtf(ms + 1e-5f) * w[t];
}

// ---------------------------------------------------------------------------
// fp32 [K,N] -> bf16 [N,K] (transpose + convert), batched via grid.z.
// ---------------------------------------------------------------------------
__global__ void cvt_t_kernel(const float* __restrict__ src, __bf16* __restrict__ dst,
                             int K, int N, size_t sStride, size_t dStride) {
    size_t z   = blockIdx.z;
    const float* s = src + z * sStride;
    __bf16*      d = dst + z * dStride;
    int idx = blockIdx.x * blockDim.x + threadIdx.x;
    if (idx < K * N) {
        int n = idx / K;
        int k = idx - n * K;
        d[(size_t)n * K + k] = (__bf16)s[(size_t)k * N + n];
    }
}

// ---------------------------------------------------------------------------
// Latent GEMM: C[32,16384] = A[32,16384] @ Wl[16384,16384] + bl.
// fp32 WMMA 16x16x4 (exact precision, no conversion traffic; Wl streamed
// exactly once: each wave owns one 16-wide N tile and both 16-row M tiles).
// global_prefetch keeps the 1 GiB Wl stream ahead of the WMMA pipe.
// ---------------------------------------------------------------------------
__global__ void latent_gemm_kernel(const float* __restrict__ A, const float* __restrict__ Bm,
                                   const float* __restrict__ bias, float* __restrict__ C) {
    const int Kd = KBIG, Nd = KBIG;
    int lane  = threadIdx.x & 31;
    int wave  = threadIdx.x >> 5;
    int n0    = (blockIdx.x * 8 + wave) * 16;
    int khalf = lane >> 4;
    int mr    = lane & 15;
    v8f acc0 = {}, acc1 = {};
#if __has_builtin(__builtin_amdgcn_wmma_f32_16x16x4_f32)
    for (int k0 = 0; k0 < Kd; k0 += 32) {
        // speculative prefetch of the B stream ~3 k-blocks ahead
        int kpf = (k0 + 96 < Kd) ? (k0 + 96) : k0;
        __builtin_prefetch(Bm + (size_t)kpf * Nd + n0 + mr, 0, 1);
#pragma unroll
        for (int kk = 0; kk < 32; kk += 4) {
            int ka = k0 + kk + 2 * khalf;
            v2f a0, a1, b;
            const float* ap0 = A + (size_t)mr * Kd + ka;
            const float* ap1 = A + (size_t)(mr + 16) * Kd + ka;
            a0.x = ap0[0]; a0.y = ap0[1];
            a1.x = ap1[0]; a1.y = ap1[1];
            const float* bp = Bm + (size_t)ka * Nd + n0 + mr;
            b.x = bp[0]; b.y = bp[Nd];
            acc0 = __builtin_amdgcn_wmma_f32_16x16x4_f32(false, a0, false, b, (short)0, acc0, false, false);
            acc1 = __builtin_amdgcn_wmma_f32_16x16x4_f32(false, a1, false, b, (short)0, acc1, false, false);
        }
    }
#else
    // Fragment-layout-identical scalar fallback (still streams Wl once).
    for (int k0 = 0; k0 < Kd; ++k0) {
        float bv = Bm[(size_t)k0 * Nd + n0 + mr];
#pragma unroll
        for (int r = 0; r < 8; ++r) {
            acc0[r] += A[(size_t)(r + 8 * khalf) * Kd + k0] * bv;
            acc1[r] += A[(size_t)(r + 8 * khalf + 16) * Kd + k0] * bv;
        }
    }
#endif
    int   cn  = n0 + mr;
    float bia = bias[cn];
#pragma unroll
    for (int r = 0; r < 8; ++r) {
        int cm = r + 8 * khalf;
        C[(size_t)cm * Nd + cn]        = acc0[r] + bia;
        C[(size_t)(cm + 16) * Nd + cn] = acc1[r] + bia;
    }
}

// ---------------------------------------------------------------------------
// Generic bf16 WMMA GEMM: C[M,N] = A[M,K] @ Bt[N,K]^T + bias (+resid)(+relu).
// 256-thread block = 8 waves; each wave register-blocks a 32x32 C tile
// (2x2 WMMAs, each A/B fragment reused twice -> 1 fragment load per wmma).
// Block tile 64(M) x 128(N); batched via grid.z.
// ---------------------------------------------------------------------------
template <bool A_BF16, bool RELU, bool OUT_BF16, bool RESADD>
__global__ void gemm_bf16_kernel(const void* __restrict__ Av, const __bf16* __restrict__ Bt,
                                 const float* __restrict__ bias, void* __restrict__ Cv,
                                 const float* __restrict__ resid, int M, int N, int Kdim,
                                 size_t aBatch, size_t bBatch, size_t biasBatch, size_t cBatch) {
    int lane = threadIdx.x & 31;
    int wave = threadIdx.x >> 5;
    int wm = wave >> 2, wn = wave & 3;
    int m0 = blockIdx.y * 64 + wm * 32;
    int n0 = blockIdx.x * 128 + wn * 32;
    if (m0 >= M || n0 >= N) return;
    size_t z = blockIdx.z;

    const float*  Af = nullptr;
    const __bf16* Ab = nullptr;
    if constexpr (A_BF16) Ab = (const __bf16*)Av + z * aBatch;
    else                  Af = (const float*)Av + z * aBatch;
    const __bf16* Bp = Bt + z * bBatch;

    int khalf = lane >> 4;
    int mrow  = m0 + (lane & 15);
    int nrow  = n0 + (lane & 15);
    v8f acc00 = {}, acc01 = {}, acc10 = {}, acc11 = {};
    for (int k0 = 0; k0 < Kdim; k0 += 32) {
        int kbA = k0 + 8 * khalf;
        int kbB = k0 + 8 * khalf;
        v16bf a0, a1;
        if constexpr (A_BF16) {
            a0 = load_a_frag_bf16(Ab, Kdim, mrow, kbA);
            a1 = load_a_frag_bf16(Ab, Kdim, mrow + 16, kbA);
        } else {
            a0 = load_a_frag_f32(Af, Kdim, mrow, kbA);
            a1 = load_a_frag_f32(Af, Kdim, mrow + 16, kbA);
        }
        v16bf b0 = load_b_frag(Bp, Kdim, nrow, kbB);
        v16bf b1 = load_b_frag(Bp, Kdim, nrow + 16, kbB);
        acc00 = __builtin_amdgcn_wmma_f32_16x16x32_bf16(false, a0, false, b0, (short)0, acc00, false, false);
        acc01 = __builtin_amdgcn_wmma_f32_16x16x32_bf16(false, a0, false, b1, (short)0, acc01, false, false);
        acc10 = __builtin_amdgcn_wmma_f32_16x16x32_bf16(false, a1, false, b0, (short)0, acc10, false, false);
        acc11 = __builtin_amdgcn_wmma_f32_16x16x32_bf16(false, a1, false, b1, (short)0, acc11, false, false);
    }
    int   cn0  = n0 + (lane & 15);
    int   cn1  = cn0 + 16;
    float bia0 = bias ? bias[z * biasBatch + cn0] : 0.0f;
    float bia1 = bias ? bias[z * biasBatch + cn1] : 0.0f;

    auto emit = [&](int cm, int cn, float v) {
        if constexpr (RESADD) v += resid[(size_t)cm * N + cn];
        if constexpr (RELU)   v = v > 0.0f ? v : 0.0f;
        if constexpr (OUT_BF16) ((__bf16*)Cv)[z * cBatch + (size_t)cm * N + cn] = (__bf16)v;
        else                    ((float*)Cv)[z * cBatch + (size_t)cm * N + cn] = v;
    };
#pragma unroll
    for (int r = 0; r < 8; ++r) {
        int cm0 = m0 + r + 8 * khalf;
        int cm1 = cm0 + 16;
        emit(cm0, cn0, acc00[r] + bia0);
        emit(cm0, cn1, acc01[r] + bia1);
        emit(cm1, cn0, acc10[r] + bia0);
        emit(cm1, cn1, acc11[r] + bia1);
    }
}

// ---------------------------------------------------------------------------
// RoPE on q [NTOK, D] (row = b*S+s, col = h*32+d), bug-faithful aliasing:
// nY uses the ALREADY-UPDATED real part.
// ---------------------------------------------------------------------------
__global__ void rope_kernel(float* __restrict__ q) {
    int idx = blockIdx.x * blockDim.x + threadIdx.x;   // NTOK * 128 pairs
    if (idx >= NTOK * (D_ / 2)) return;
    int row  = idx >> 7;          // token
    int pair = idx & 127;         // h*16 + i
    int h = pair >> 4, i = pair & 15;
    int s = row & (S_ - 1);
    float freq = powf(10000.0f, -2.0f * (float)i / (float)HD_);
    float ang  = (float)s * freq;
    float c = cosf(ang), sn = sinf(ang);
    float* p = q + (size_t)row * D_ + h * HD_ + 2 * i;
    float X = p[0], Y = p[1];
    float nX = X * c - Y * sn;
    float nY = nX * sn + Y * c;   // bug-faithful: uses updated X
    p[0] = nX; p[1] = nY;
}

// ---------------------------------------------------------------------------
// Attention core: one block per (b,h), 64 threads (thread = query position).
// q,k,v are [NTOK, D]; head h occupies columns h*32..h*32+31. L = S = 64.
// ---------------------------------------------------------------------------
__global__ void attn_kernel(const float* __restrict__ q, const float* __restrict__ k,
                            const float* __restrict__ v, float* __restrict__ o) {
    int b = blockIdx.x >> 3;
    int h = blockIdx.x & 7;
    int s = threadIdx.x;          // 0..63
    __shared__ float ks[S_][HD_];
    __shared__ float vs[S_][HD_];
    {
        const float* kp = k + ((size_t)(b * S_ + s)) * D_ + h * HD_;
        const float* vp = v + ((size_t)(b * S_ + s)) * D_ + h * HD_;
#pragma unroll
        for (int d = 0; d < HD_; ++d) { ks[s][d] = kp[d]; vs[s][d] = vp[d]; }
    }
    __syncthreads();
    float qr[HD_];
    const float* qp = q + ((size_t)(b * S_ + s)) * D_ + h * HD_;
#pragma unroll
    for (int d = 0; d < HD_; ++d) qr[d] = qp[d];

    const float scale = 0.17677669529663688f;   // 1/sqrt(32)
    float sc[S_];
    float mx = -3.4e38f;
    for (int l = 0; l < S_; ++l) {
        float a = 0.0f;
#pragma unroll
        for (int d = 0; d < HD_; ++d) a += qr[d] * ks[l][d];
        a *= scale;
        sc[l] = a;
        mx = a > mx ? a : mx;
    }
    float sum = 0.0f;
    for (int l = 0; l < S_; ++l) { sc[l] = expf(sc[l] - mx); sum += sc[l]; }
    float inv = 1.0f / sum;
    float outv[HD_];
#pragma unroll
    for (int d = 0; d < HD_; ++d) outv[d] = 0.0f;
    for (int l = 0; l < S_; ++l) {
        float p = sc[l] * inv;
#pragma unroll
        for (int d = 0; d < HD_; ++d) outv[d] += p * vs[l][d];
    }
    float* op = o + ((size_t)(b * S_ + s)) * D_ + h * HD_;
#pragma unroll
    for (int d = 0; d < HD_; ++d) op[d] = outv[d];
}

// ---------------------------------------------------------------------------
// Router: one wave per token. logits = x@rw + rb, softmax(E=8), top-2 gates.
// gates[t][e] = prob if e in top2 else 0.
// ---------------------------------------------------------------------------
__global__ void router_kernel(const float* __restrict__ xn, const float* __restrict__ rw,
                              const float* __restrict__ rb, float* __restrict__ gates) {
    int gtid = blockIdx.x * blockDim.x + threadIdx.x;
    int tok  = gtid >> 5;
    int lane = threadIdx.x & 31;
    if (tok >= NTOK) return;
    const float* xp = xn + (size_t)tok * D_;
    float logit[E_];
#pragma unroll
    for (int e = 0; e < E_; ++e) {
        float acc = 0.0f;
        for (int d = lane; d < D_; d += 32) acc += xp[d] * rw[d * E_ + e];
#pragma unroll
        for (int off = 16; off > 0; off >>= 1) acc += __shfl_xor(acc, off, 32);
        logit[e] = acc + rb[e];
    }
    if (lane == 0) {
        float mx = logit[0];
#pragma unroll
        for (int e = 1; e < E_; ++e) mx = logit[e] > mx ? logit[e] : mx;
        float p[E_], sum = 0.0f;
#pragma unroll
        for (int e = 0; e < E_; ++e) { p[e] = expf(logit[e] - mx); sum += p[e]; }
        float inv = 1.0f / sum;
#pragma unroll
        for (int e = 0; e < E_; ++e) p[e] *= inv;
        int i1 = 0;
#pragma unroll
        for (int e = 1; e < E_; ++e) if (p[e] > p[i1]) i1 = e;
        int i2 = (i1 == 0) ? 1 : 0;
#pragma unroll
        for (int e = 0; e < E_; ++e) if (e != i1 && p[e] > p[i2]) i2 = e;
#pragma unroll
        for (int e = 0; e < E_; ++e)
            gates[(size_t)tok * E_ + e] = (e == i1 || e == i2) ? p[e] : 0.0f;
    }
}

// ---------------------------------------------------------------------------
// Fused MoE output: out[t,d] = x1[t,d] + sum_e gate[t,e]*(h2[e]@W2[e] + b2[e]).
// Expert loop inside the tile (no eo tensor, no atomics, deterministic).
// Each wave register-blocks 32x32; block tile 64 x 128.
// ---------------------------------------------------------------------------
__global__ void moe_out_kernel(const __bf16* __restrict__ h2, const __bf16* __restrict__ W2t,
                               const float* __restrict__ b2, const float* __restrict__ gates,
                               const float* __restrict__ resid, float* __restrict__ out) {
    int lane = threadIdx.x & 31;
    int wave = threadIdx.x >> 5;
    int wm = wave >> 2, wn = wave & 3;
    int m0 = blockIdx.y * 64 + wm * 32;
    int n0 = blockIdx.x * 128 + wn * 32;
    int khalf = lane >> 4;
    int mrow  = m0 + (lane & 15);
    int nrow  = n0 + (lane & 15);
    int cn0   = nrow;
    int cn1   = nrow + 16;

    float res00[8], res01[8], res10[8], res11[8];
#pragma unroll
    for (int r = 0; r < 8; ++r) {
        int cm0 = m0 + r + 8 * khalf;
        int cm1 = cm0 + 16;
        res00[r] = resid[(size_t)cm0 * D_ + cn0];
        res01[r] = resid[(size_t)cm0 * D_ + cn1];
        res10[r] = resid[(size_t)cm1 * D_ + cn0];
        res11[r] = resid[(size_t)cm1 * D_ + cn1];
    }

    for (int e = 0; e < E_; ++e) {
        const __bf16* Ae = h2  + (size_t)e * NTOK * F_;
        const __bf16* Be = W2t + (size_t)e * D_ * F_;
        v8f acc00 = {}, acc01 = {}, acc10 = {}, acc11 = {};
        for (int k0 = 0; k0 < F_; k0 += 32) {
            int kbA = k0 + 8 * khalf;
            int kbB = k0 + 8 * khalf;
            v16bf a0 = load_a_frag_bf16(Ae, F_, mrow, kbA);
            v16bf a1 = load_a_frag_bf16(Ae, F_, mrow + 16, kbA);
            v16bf b0 = load_b_frag(Be, F_, cn0, kbB);
            v16bf b1 = load_b_frag(Be, F_, cn1, kbB);
            acc00 = __builtin_amdgcn_wmma_f32_16x16x32_bf16(false, a0, false, b0, (short)0, acc00, false, false);
            acc01 = __builtin_amdgcn_wmma_f32_16x16x32_bf16(false, a0, false, b1, (short)0, acc01, false, false);
            acc10 = __builtin_amdgcn_wmma_f32_16x16x32_bf16(false, a1, false, b0, (short)0, acc10, false, false);
            acc11 = __builtin_amdgcn_wmma_f32_16x16x32_bf16(false, a1, false, b1, (short)0, acc11, false, false);
        }
        float bia0 = b2[e * D_ + cn0];
        float bia1 = b2[e * D_ + cn1];
#pragma unroll
        for (int r = 0; r < 8; ++r) {
            int cm0 = m0 + r + 8 * khalf;
            int cm1 = cm0 + 16;
            float g0 = gates[(size_t)cm0 * E_ + e];
            float g1 = gates[(size_t)cm1 * E_ + e];
            res00[r] += g0 * (acc00[r] + bia0);
            res01[r] += g0 * (acc01[r] + bia1);
            res10[r] += g1 * (acc10[r] + bia0);
            res11[r] += g1 * (acc11[r] + bia1);
        }
    }
#pragma unroll
    for (int r = 0; r < 8; ++r) {
        int cm0 = m0 + r + 8 * khalf;
        int cm1 = cm0 + 16;
        out[(size_t)cm0 * D_ + cn0] = res00[r];
        out[(size_t)cm0 * D_ + cn1] = res01[r];
        out[(size_t)cm1 * D_ + cn0] = res10[r];
        out[(size_t)cm1 * D_ + cn1] = res11[r];
    }
}

// ---------------------------------------------------------------------------
// Host orchestration
// ---------------------------------------------------------------------------
extern "C" void kernel_launch(void* const* d_in, const int* in_sizes, int n_in,
                              void* d_out, int out_size, void* d_ws, size_t ws_size,
                              hipStream_t stream) {
    (void)in_sizes; (void)n_in; (void)out_size; (void)ws_size;
    const float* x    = (const float*)d_in[0];
    const float* rms1 = (const float*)d_in[1];
    const float* Wl   = (const float*)d_in[2];
    const float* bl   = (const float*)d_in[3];
    const float* Wq   = (const float*)d_in[4];
    const float* bq   = (const float*)d_in[5];
    const float* Wk   = (const float*)d_in[6];
    const float* bk   = (const float*)d_in[7];
    const float* Wv   = (const float*)d_in[8];
    const float* bv   = (const float*)d_in[9];
    const float* Wo   = (const float*)d_in[10];
    const float* bo   = (const float*)d_in[11];
    const float* rms2 = (const float*)d_in[12];
    const float* rw   = (const float*)d_in[13];
    const float* rb   = (const float*)d_in[14];
    const float* W1   = (const float*)d_in[15];
    const float* b1   = (const float*)d_in[16];
    const float* Wsw  = (const float*)d_in[17];
    const float* bs   = (const float*)d_in[18];
    const float* W2   = (const float*)d_in[19];
    const float* b2   = (const float*)d_in[20];
    float* out = (float*)d_out;

    char* ws = (char*)d_ws;
    size_t off = 0;
    auto alloc = [&](size_t bytes) -> void* {
        void* p = ws + off;
        off += (bytes + 255) & ~(size_t)255;
        return p;
    };
    float*  xn1   = (float*)alloc((size_t)NTOK * D_ * 4);
    float*  lat   = (float*)alloc((size_t)B_ * LAT_ * D_ * 4);
    float*  qp    = (float*)alloc((size_t)NTOK * D_ * 4);
    float*  kp    = (float*)alloc((size_t)NTOK * D_ * 4);
    float*  vp    = (float*)alloc((size_t)NTOK * D_ * 4);
    float*  attn  = (float*)alloc((size_t)NTOK * D_ * 4);
    float*  x1    = (float*)alloc((size_t)NTOK * D_ * 4);
    float*  xn2   = (float*)alloc((size_t)NTOK * D_ * 4);
    float*  gates = (float*)alloc((size_t)NTOK * E_ * 4);
    __bf16* Wqt   = (__bf16*)alloc((size_t)D_ * D_ * 2);
    __bf16* Wkt   = (__bf16*)alloc((size_t)D_ * D_ * 2);
    __bf16* Wvt   = (__bf16*)alloc((size_t)D_ * D_ * 2);
    __bf16* Wot   = (__bf16*)alloc((size_t)D_ * D_ * 2);
    __bf16* W1t   = (__bf16*)alloc((size_t)E_ * F_ * D_ * 2);
    __bf16* Wst   = (__bf16*)alloc((size_t)E_ * F_ * F_ * 2);
    __bf16* W2t   = (__bf16*)alloc((size_t)E_ * D_ * F_ * 2);
    __bf16* h1b   = (__bf16*)alloc((size_t)E_ * NTOK * F_ * 2);
    __bf16* h2b   = (__bf16*)alloc((size_t)E_ * NTOK * F_ * 2);

    // 1. xn1 = rmsnorm(x, rms1_w)
    rmsnorm_kernel<<<NTOK, 256, 0, stream>>>(x, rms1, xn1);

    // 2. Weight convert + transpose to bf16 [N,K] (reused many times -> worth it)
    auto cvtT = [&](const float* src, __bf16* dst, int K, int N, int batch) {
        size_t total = (size_t)K * N;
        dim3 g((unsigned)((total + 255) / 256), 1, (unsigned)batch);
        cvt_t_kernel<<<g, 256, 0, stream>>>(src, dst, K, N, total, total);
    };
    cvtT(Wq, Wqt, D_, D_, 1);
    cvtT(Wk, Wkt, D_, D_, 1);
    cvtT(Wv, Wvt, D_, D_, 1);
    cvtT(Wo, Wot, D_, D_, 1);
    cvtT(W1, W1t, D_, F_, E_);
    cvtT(Wsw, Wst, F_, F_, E_);
    cvtT(W2, W2t, F_, D_, E_);

    // 3. lat[32,16384] = xn1.reshape(32,16384) @ Wl + bl  (fp32 WMMA, Wl streamed once)
    latent_gemm_kernel<<<KBIG / 128, 256, 0, stream>>>(xn1, Wl, bl, lat);

    // 4. q/k/v projections (lat reinterpreted as [NTOK, D]); block tile 64x128
    dim3 gP(D_ / 128, NTOK / 64, 1);   // (2, 32)
    gemm_bf16_kernel<false, false, false, false><<<gP, 256, 0, stream>>>(
        xn1, Wqt, bq, qp, nullptr, NTOK, D_, D_, 0, 0, 0, 0);
    gemm_bf16_kernel<false, false, false, false><<<gP, 256, 0, stream>>>(
        lat, Wkt, bk, kp, nullptr, NTOK, D_, D_, 0, 0, 0, 0);
    gemm_bf16_kernel<false, false, false, false><<<gP, 256, 0, stream>>>(
        lat, Wvt, bv, vp, nullptr, NTOK, D_, D_, 0, 0, 0, 0);

    // 5. RoPE on q (bug-faithful)
    rope_kernel<<<(NTOK * (D_ / 2)) / 256, 256, 0, stream>>>(qp);

    // 6. attention core
    attn_kernel<<<B_ * H_, S_, 0, stream>>>(qp, kp, vp, attn);

    // 7. x1 = x + attn @ Wo + bo (residual fused)
    gemm_bf16_kernel<false, false, false, true><<<gP, 256, 0, stream>>>(
        attn, Wot, bo, x1, x, NTOK, D_, D_, 0, 0, 0, 0);

    // 8. xn2 = rmsnorm(x1, rms2_w)
    rmsnorm_kernel<<<NTOK, 256, 0, stream>>>(x1, rms2, xn2);

    // 9. router gates (softmax + top-2)
    router_kernel<<<NTOK / 8, 256, 0, stream>>>(xn2, rw, rb, gates);

    // 10. h1[e] = xn2 @ W1[e] + b1[e]   (out bf16)
    dim3 gF(F_ / 128, NTOK / 64, E_);  // (8, 32, 8)
    gemm_bf16_kernel<false, false, true, false><<<gF, 256, 0, stream>>>(
        xn2, W1t, b1, h1b, nullptr, NTOK, F_, D_,
        0, (size_t)F_ * D_, (size_t)F_, (size_t)NTOK * F_);

    // 11. h2[e] = relu(h1[e] @ Ws[e] + bs[e])   (bf16 in/out)
    gemm_bf16_kernel<true, true, true, false><<<gF, 256, 0, stream>>>(
        h1b, Wst, bs, h2b, nullptr, NTOK, F_, F_,
        (size_t)NTOK * F_, (size_t)F_ * F_, (size_t)F_, (size_t)NTOK * F_);

    // 12. out = x1 + sum_e gate[t,e]*(h2[e] @ W2[e] + b2[e])
    dim3 gO(D_ / 128, NTOK / 64, 1);   // (2, 32)
    moe_out_kernel<<<gO, 256, 0, stream>>>(h2b, W2t, b2, gates, x1, out);
}
